// SupervisedGraphSage_4011499454939
// MI455X (gfx1250) — matrix operand order repaired and checked by reading, not compiled, not run on hardware
//
#include <hip/hip_runtime.h>
#include <hip/hip_fp16.h>

typedef __attribute__((ext_vector_type(16))) _Float16 v16h;
typedef __attribute__((ext_vector_type(8)))  _Float16 v8h;
typedef __attribute__((ext_vector_type(8)))  float    v8f;

#define N_NODES 50000
#define S_NEIGH 10

// ---------------- weight f32 -> f16 (with optional zero pad tail) ----------------
__global__ void wconv_kernel(const float* __restrict__ src, _Float16* __restrict__ dst,
                             int n_src, int n_total) {
  int i = blockIdx.x * blockDim.x + threadIdx.x;
  if (i < n_total) dst[i] = (i < n_src) ? (_Float16)src[i] : (_Float16)0.0f;
}

// ---------------- layer 1: K=6 is tiny, plain VALU kernel ----------------
// h1[n][e] = relu( sum_j<3 raw[n][j]*W1[e][j] + mean_s(raw[neigh][j])*W1[e][3+j] )
__global__ void layer1_kernel(const float* __restrict__ raw,
                              const int* __restrict__ neigh,
                              const float* __restrict__ W1,
                              _Float16* __restrict__ h1) {
  __shared__ float s[6];
  const int n = blockIdx.x;
  const int t = threadIdx.x;               // 256 threads = 256 outputs
  if (t < 6) {
    if (t < 3) {
      s[t] = raw[n * 3 + t];
    } else {
      const int j = t - 3;
      float acc = 0.0f;
      for (int k = 0; k < S_NEIGH; ++k)
        acc += raw[(size_t)neigh[n * S_NEIGH + k] * 3 + j];
      s[t] = acc * (1.0f / S_NEIGH);
    }
  }
  __syncthreads();
  const float f0 = s[0], f1 = s[1], f2 = s[2], f3 = s[3], f4 = s[4], f5 = s[5];
  const float* w = W1 + t * 6;
  float acc = f0 * w[0] + f1 * w[1] + f2 * w[2] + f3 * w[3] + f4 * w[4] + f5 * w[5];
  h1[(size_t)n * 256 + t] = (_Float16)fmaxf(acc, 0.0f);
}

// ---------------- gather/mean: comb[n] = [ h[n] , mean_s h[neigh[n,s]] ] (f16) ----
// one node per block (64 threads), v8h (16B) vector loads; h fits in 192MB L2.
template <int D>
__global__ void gather_kernel(const _Float16* __restrict__ h,
                              const int* __restrict__ neigh,
                              _Float16* __restrict__ comb) {
  const int n = blockIdx.x;
  const int t = threadIdx.x;              // 64 threads
  constexpr int CH = D / 8;               // 8 halves per chunk
  if (t >= CH) return;
  int idx[S_NEIGH];
#pragma unroll
  for (int s = 0; s < S_NEIGH; ++s) idx[s] = neigh[n * S_NEIGH + s];

  float sum[8] = {};
#pragma unroll
  for (int s = 0; s < S_NEIGH; ++s) {
    v8h x = *(const v8h*)(h + (size_t)idx[s] * D + t * 8);
#pragma unroll
    for (int j = 0; j < 8; ++j) sum[j] += (float)x[j];
  }
  v8h self = *(const v8h*)(h + (size_t)n * D + t * 8);
  v8h mean;
#pragma unroll
  for (int j = 0; j < 8; ++j) mean[j] = (_Float16)(sum[j] * (1.0f / S_NEIGH));

  *(v8h*)(comb + (size_t)n * (2 * D) + t * 8)     = self;
  *(v8h*)(comb + (size_t)n * (2 * D) + D + t * 8) = mean;
}

// ---------------- WMMA GEMM:  out = [relu](A[M,K] @ B[Nout,K]^T) ----------------
// 4 waves/block; each wave computes a 16 x (NT*16) strip via v_wmma_f32_16x16x32_f16.
// A fragment (16-bit 16x32): lane<16 -> K kb..kb+15 contiguous; lane>=16 -> kb+16..kb+31.
// B fragment (16-bit 32x16): lane<16 -> K kb..kb+15 contiguous; lane>=16 -> kb+16..kb+31.
template <int NT, bool RELU_F16>
__launch_bounds__(128)
__global__ void wmma_gemm_kernel(const _Float16* __restrict__ A,
                                 const _Float16* __restrict__ B,
                                 int M, int K,
                                 _Float16* __restrict__ outH, int ldoH,
                                 float* __restrict__ outF, int ldoF, int colMax) {
  const int lane = threadIdx.x & 31;
  const int wave = threadIdx.x >> 5;
  const int row0 = (blockIdx.x * (int)(blockDim.x >> 5) + wave) * 16;
  if (row0 >= M) return;                  // wave-uniform: EXEC stays all-ones
  const int col0 = blockIdx.y * (NT * 16);

  const int m  = lane & 15;
  const int kl = (lane < 16) ? 0 : 16;    // A lane K offset
  const int bo = (lane < 16) ? 0 : 16;    // B lane K offset

  const _Float16* arow = A + (size_t)(row0 + m) * K;
  const _Float16* brow[NT];
#pragma unroll
  for (int t = 0; t < NT; ++t) brow[t] = B + (size_t)(col0 + t * 16 + m) * K;

  v8f acc[NT];
#pragma unroll
  for (int t = 0; t < NT; ++t) acc[t] = (v8f){};

  for (int kb = 0; kb < K; kb += 32) {
    union { v16h v; v8h h[2]; } a;
    a.h[0] = *(const v8h*)(arow + kb + kl);
    a.h[1] = *(const v8h*)(arow + kb + kl + 8);
#pragma unroll
    for (int t = 0; t < NT; ++t) {
      v16h b = *(const v16h*)(brow[t] + kb + bo);
      acc[t] = __builtin_amdgcn_wmma_f32_16x16x32_f16(
          /*neg_a=*/false, a.v, /*neg_b=*/false, b,
          /*c_mod=*/(short)0, acc[t], /*reuse_a=*/false, /*reuse_b=*/false);
    }
  }

  // D layout: VGPR i -> row = row0 + i + (lane>=16 ? 8 : 0), col = col0 + t*16 + (lane&15)
#pragma unroll
  for (int t = 0; t < NT; ++t) {
#pragma unroll
    for (int i = 0; i < 8; ++i) {
      const int row = row0 + i + ((lane >= 16) ? 8 : 0);
      const int col = col0 + t * 16 + m;
      if constexpr (RELU_F16) {
        outH[(size_t)row * ldoH + col] = (_Float16)fmaxf(acc[t][i], 0.0f);
      } else {
        if (col < colMax) outF[(size_t)row * ldoF + col] = acc[t][i];
      }
    }
  }
}

extern "C" void kernel_launch(void* const* d_in, const int* in_sizes, int n_in,
                              void* d_out, int out_size, void* d_ws, size_t ws_size,
                              hipStream_t stream) {
  const float* raw   = (const float*)d_in[0];
  const int*   neigh = (const int*)d_in[1];
  // d_in[2] = nodes = arange(N): identity gather, ignored.
  const float* W1 = (const float*)d_in[3];
  const float* W2 = (const float*)d_in[4];
  const float* W3 = (const float*)d_in[5];
  const float* W4 = (const float*)d_in[6];
  const float* Wc = (const float*)d_in[7];
  float* out = (float*)d_out;

  char* ws = (char*)d_ws;
  _Float16* hbuf = (_Float16*)ws;                                   // N x 512 f16
  _Float16* cbuf = (_Float16*)(ws + (size_t)N_NODES * 512 * 2);     // N x 1024 f16
  _Float16* wbuf = (_Float16*)(ws + (size_t)N_NODES * 512 * 2
                                  + (size_t)N_NODES * 1024 * 2);    // 512x1024 f16 max

  const int MT = (N_NODES / 16 + 3) / 4;     // 3125 row-tiles, 4 waves/block -> 782
  const dim3 gHid(MT, 512 / 64);             // NT=4 -> 64 cols per block.y
  const dim3 gCls(MT, 1);                    // NT=3 -> 48 padded cols

  // ---- layer 1: raw[N,3] -> h1[N,256] f16 (stride 256) ----
  layer1_kernel<<<N_NODES, 256, 0, stream>>>(raw, neigh, W1, hbuf);

  // ---- layer 2: K=512, Nout=512 ----
  gather_kernel<256><<<N_NODES, 64, 0, stream>>>(hbuf, neigh, cbuf);
  wconv_kernel<<<(512 * 512 + 255) / 256, 256, 0, stream>>>(W2, wbuf, 512 * 512, 512 * 512);
  wmma_gemm_kernel<4, true><<<gHid, 128, 0, stream>>>(cbuf, wbuf, N_NODES, 512,
                                                      hbuf, 512, nullptr, 0, 0);

  // ---- layer 3: K=1024, Nout=512 ----
  gather_kernel<512><<<N_NODES, 64, 0, stream>>>(hbuf, neigh, cbuf);
  wconv_kernel<<<(512 * 1024 + 255) / 256, 256, 0, stream>>>(W3, wbuf, 512 * 1024, 512 * 1024);
  wmma_gemm_kernel<4, true><<<gHid, 128, 0, stream>>>(cbuf, wbuf, N_NODES, 1024,
                                                      hbuf, 512, nullptr, 0, 0);

  // ---- layer 4: K=1024, Nout=512 ----
  gather_kernel<512><<<N_NODES, 64, 0, stream>>>(hbuf, neigh, cbuf);
  wconv_kernel<<<(512 * 1024 + 255) / 256, 256, 0, stream>>>(W4, wbuf, 512 * 1024, 512 * 1024);
  wmma_gemm_kernel<4, true><<<gHid, 128, 0, stream>>>(cbuf, wbuf, N_NODES, 1024,
                                                      hbuf, 512, nullptr, 0, 0);

  // ---- classifier: out[N,40] = h4 @ Wc^T  (Nout padded to 48, no relu, f32 out) ----
  wconv_kernel<<<(48 * 512 + 255) / 256, 256, 0, stream>>>(Wc, wbuf, 40 * 512, 48 * 512);
  wmma_gemm_kernel<3, false><<<gCls, 128, 0, stream>>>(hbuf, wbuf, N_NODES, 512,
                                                       nullptr, 0, out, 40, 40);
}
